// Router_49203145343605
// MI455X (gfx1250) — matrix operand order repaired and checked by reading, not compiled, not run on hardware
//
#include <hip/hip_runtime.h>
#include <hip/hip_bf16.h>
#include <math.h>

// Problem constants (fixed by the reference): B=4, S=4096, H=2048, Hm=1024, E=8
#define ROWS   16384      // B*S
#define KDIM   2048       // H
#define NDIM   1024       // H/2
#define NEXP   8
#define BM     128
#define BN     128
#define BK     32
#define LDST   40         // LDS row pitch in bf16 halves (80B = 20 banks -> conflict-free)

typedef __attribute__((ext_vector_type(16))) __bf16 v16bf;
typedef __attribute__((ext_vector_type(8)))  __bf16 v8bf;
typedef __attribute__((ext_vector_type(8)))  float  v8f;

// Registers holding one block-tile of X (128x32) and W1 (32x128) in flight.
struct TileRegs {
    float4 x[4];
    float4 w[4];
};

__device__ __forceinline__ void tiles_load(TileRegs& t,
                                           const float* __restrict__ X,
                                           const float* __restrict__ W1,
                                           int m0, int n0, int k0, int tid)
{
#pragma unroll
    for (int it = 0; it < 4; ++it) {
        const int g = tid + it * 256;                 // 1024 float4 groups
        const float* a = X + (size_t)(m0 + (g >> 3)) * KDIM + k0 + (g & 7) * 4;
        t.x[it] = *(const float4*)a;
        if (k0 + BK < KDIM) __builtin_prefetch(a + BK, 0, 0);       // tile k+2
    }
#pragma unroll
    for (int it = 0; it < 4; ++it) {
        const int g = tid + it * 256;
        const float* a = W1 + (size_t)(k0 + (g >> 5)) * NDIM + n0 + (g & 31) * 4;
        t.w[it] = *(const float4*)a;
        if (k0 + BK < KDIM) __builtin_prefetch(a + (size_t)BK * NDIM, 0, 0);
    }
}

__device__ __forceinline__ void tiles_store(const TileRegs& t, int tid,
                                            __bf16* __restrict__ s_xh,
                                            __bf16* __restrict__ s_xl,
                                            __bf16* __restrict__ s_wh,
                                            __bf16* __restrict__ s_wl)
{
#pragma unroll
    for (int it = 0; it < 4; ++it) {
        const int g  = tid + it * 256;
        const int r  = g >> 3;                        // 0..127
        const int kq = (g & 7) * 4;                   // 0,4,..,28
        const float f[4] = {t.x[it].x, t.x[it].y, t.x[it].z, t.x[it].w};
#pragma unroll
        for (int i = 0; i < 4; ++i) {
            const __bf16 hi = (__bf16)f[i];
            s_xh[r * LDST + kq + i] = hi;
            s_xl[r * LDST + kq + i] = (__bf16)(f[i] - (float)hi);
        }
    }
#pragma unroll
    for (int it = 0; it < 4; ++it) {
        const int g  = tid + it * 256;
        const int kk = g >> 5;                        // 0..31
        const int nq = (g & 31) * 4;                  // 0,4,..,124
        const float f[4] = {t.w[it].x, t.w[it].y, t.w[it].z, t.w[it].w};
#pragma unroll
        for (int i = 0; i < 4; ++i) {
            const __bf16 hi = (__bf16)f[i];
            s_wh[(nq + i) * LDST + kk] = hi;          // transposed [n][k]
            s_wl[(nq + i) * LDST + kk] = (__bf16)(f[i] - (float)hi);
        }
    }
}

// ---------------------------------------------------------------------------
// Kernel 1: h = relu(X @ W1 + b1), fp32 via bf16x3 split-precision WMMA.
// Block = 256 threads = 8 wave32, tile 128x128, K-step 32, double-buffered LDS
// software pipeline: next tile's global loads issue before the WMMA block so
// HBM latency hides behind the 24 v_wmma per step.
// ---------------------------------------------------------------------------
__global__ __launch_bounds__(256)
void router_gemm1(const float* __restrict__ X,
                  const float* __restrict__ W1,
                  const float* __restrict__ b1,
                  float* __restrict__ Hout)
{
    __shared__ __align__(16) __bf16 s_xh[2][BM * LDST];
    __shared__ __align__(16) __bf16 s_xl[2][BM * LDST];
    __shared__ __align__(16) __bf16 s_wh[2][BN * LDST];  // W1 tile, transposed [n][k]
    __shared__ __align__(16) __bf16 s_wl[2][BN * LDST];

    const int tid  = threadIdx.x;
    const int lane = tid & 31;
    const int wave = tid >> 5;
    const int wm   = wave >> 1;        // 0..3  -> 32-row strip
    const int wn   = wave & 1;         // 0..1  -> 64-col strip
    const int m0   = blockIdx.y * BM;
    const int n0   = blockIdx.x * BN;
    const int hlf  = lane >> 4;        // lane half (wave32 WMMA layout)
    const int l15  = lane & 15;

    const v8f zero = {0.f, 0.f, 0.f, 0.f, 0.f, 0.f, 0.f, 0.f};
    v8f acc[2][4];
#pragma unroll
    for (int i = 0; i < 2; ++i)
#pragma unroll
        for (int j = 0; j < 4; ++j) acc[i][j] = zero;

    // ---- pipeline prologue: stage tile 0 into buffer 0 ---------------------
    TileRegs t;
    tiles_load(t, X, W1, m0, n0, 0, tid);
    tiles_store(t, tid, s_xh[0], s_xl[0], s_wh[0], s_wl[0]);
    __syncthreads();

    const int NSTEP = KDIM / BK;       // 64
    for (int ks = 0; ks < NSTEP; ++ks) {
        const int  p    = ks & 1;
        const bool more = (ks + 1) < NSTEP;

        // issue next tile's global loads now; s_wait lands after the WMMAs
        if (more) tiles_load(t, X, W1, m0, n0, (ks + 1) * BK, tid);

        const __bf16* xh = s_xh[p];
        const __bf16* xl = s_xl[p];
        const __bf16* wh = s_wh[p];
        const __bf16* wl = s_wl[p];

        // ---- fragments per documented wave32 WMMA layouts ------------------
        // A 16x32 bf16: lanes 0-15 hold K 0-7 & 16-23; lanes 16-31 K 8-15 & 24-31.
        v16bf ah[2], al[2];
#pragma unroll
        for (int i = 0; i < 2; ++i) {
            const int row = wm * 32 + i * 16 + l15;
            const v8bf* ph = (const v8bf*)(xh + row * LDST + hlf * 8);
            const v8bf* pl = (const v8bf*)(xl + row * LDST + hlf * 8);
            const v8bf h0 = ph[0], h1 = ph[2];       // +16 halves
            const v8bf c0 = pl[0], c1 = pl[2];
#pragma unroll
            for (int q = 0; q < 8; ++q) {
                ah[i][q] = h0[q]; ah[i][8 + q] = h1[q];
                al[i][q] = c0[q]; al[i][8 + q] = c1[q];
            }
        }
        // B 32x16 bf16: lanes 0-15 hold K 0-7 & 16-23; lanes 16-31 K 8-15 & 24-31 (col = lane&15).
        v16bf bh[4], bl[4];
#pragma unroll
        for (int j = 0; j < 4; ++j) {
            const int col = wn * 64 + j * 16 + l15;
            const v8bf* ph = (const v8bf*)(wh + col * LDST + hlf * 8);
            const v8bf* pl = (const v8bf*)(wl + col * LDST + hlf * 8);
            const v8bf h0 = ph[0], h1 = ph[2];
            const v8bf c0 = pl[0], c1 = pl[2];
#pragma unroll
            for (int q = 0; q < 8; ++q) {
                bh[j][q] = h0[q]; bh[j][8 + q] = h1[q];
                bl[j][q] = c0[q]; bl[j][8 + q] = c1[q];
            }
        }

        // ---- bf16x3 split-precision MMA: A*B ~= AhBh + AlBh + AhBl ---------
#pragma unroll
        for (int i = 0; i < 2; ++i)
#pragma unroll
            for (int j = 0; j < 4; ++j) {
                acc[i][j] = __builtin_amdgcn_wmma_f32_16x16x32_bf16(
                    false, ah[i], false, bh[j], (short)0, acc[i][j], false, false);
                acc[i][j] = __builtin_amdgcn_wmma_f32_16x16x32_bf16(
                    false, al[i], false, bh[j], (short)0, acc[i][j], false, false);
                acc[i][j] = __builtin_amdgcn_wmma_f32_16x16x32_bf16(
                    false, ah[i], false, bl[j], (short)0, acc[i][j], false, false);
            }

        // ---- convert + store next tile into the other buffer ---------------
        if (more) tiles_store(t, tid, s_xh[p ^ 1], s_xl[p ^ 1], s_wh[p ^ 1], s_wl[p ^ 1]);
        __syncthreads();   // readers of buf p done; buf p^1 fully written
    }

    // ---- epilogue: bias + ReLU, C layout: vgpr r -> M = hlf*8 + r, N = l15 -
#pragma unroll
    for (int i = 0; i < 2; ++i)
#pragma unroll
        for (int j = 0; j < 4; ++j) {
            const int col  = n0 + wn * 64 + j * 16 + l15;
            const float bv = b1[col];
#pragma unroll
            for (int r = 0; r < 8; ++r) {
                const int row = m0 + wm * 32 + i * 16 + hlf * 8 + r;
                float v = acc[i][j][r] + bv;
                Hout[(size_t)row * NDIM + col] = v > 0.f ? v : 0.f;
            }
        }
}

// ---------------------------------------------------------------------------
// Kernel 2: logits = h @ W2 + b2, softmax, top-2. One wave32 per token row.
// ---------------------------------------------------------------------------
__global__ __launch_bounds__(256)
void router_stage2(const float* __restrict__ Hin,
                   const float* __restrict__ W2,
                   const float* __restrict__ b2,
                   float* __restrict__ out)
{
    __shared__ float w2t[NEXP * NDIM];   // W2 transposed: [e][k], 32KB

    const int tid  = threadIdx.x;
    const int lane = tid & 31;
    const int wave = tid >> 5;

#pragma unroll
    for (int it = 0; it < 32; ++it) {
        const int idx = tid + it * 256;          // 0..8191, coalesced read
        w2t[(idx & 7) * NDIM + (idx >> 3)] = W2[idx];
    }
    __syncthreads();

    const int row = blockIdx.x * 8 + wave;
    const float* hrow = Hin + (size_t)row * NDIM;

    float part[NEXP] = {0.f, 0.f, 0.f, 0.f, 0.f, 0.f, 0.f, 0.f};
    for (int j = 0; j < NDIM / 32; ++j) {
        const float hv = hrow[j * 32 + lane];
#pragma unroll
        for (int e = 0; e < NEXP; ++e)
            part[e] = fmaf(hv, w2t[e * NDIM + j * 32 + lane], part[e]);
    }
#pragma unroll
    for (int e = 0; e < NEXP; ++e) {
        part[e] += __shfl_xor(part[e], 16, 32);
        part[e] += __shfl_xor(part[e], 8,  32);
        part[e] += __shfl_xor(part[e], 4,  32);
        part[e] += __shfl_xor(part[e], 2,  32);
        part[e] += __shfl_xor(part[e], 1,  32);
        part[e] += b2[e];
    }

    if (lane == 0) {
        // logits
        float* lg = out + (size_t)row * NEXP;
#pragma unroll
        for (int e = 0; e < NEXP; ++e) lg[e] = part[e];
        // softmax
        float mx = part[0];
#pragma unroll
        for (int e = 1; e < NEXP; ++e) mx = fmaxf(mx, part[e]);
        float p[NEXP], s = 0.f;
#pragma unroll
        for (int e = 0; e < NEXP; ++e) { p[e] = expf(part[e] - mx); s += p[e]; }
        const float inv = 1.0f / s;
#pragma unroll
        for (int e = 0; e < NEXP; ++e) p[e] *= inv;
        // top-2, lowest index wins ties (strict >) to match jax.lax.top_k
        int i0 = 0; float v0 = p[0];
#pragma unroll
        for (int e = 1; e < NEXP; ++e) if (p[e] > v0) { v0 = p[e]; i0 = e; }
        int i1 = -1; float v1 = -1e30f;
#pragma unroll
        for (int e = 0; e < NEXP; ++e) if (e != i0 && p[e] > v1) { v1 = p[e]; i1 = e; }

        float* wts = out + (size_t)ROWS * NEXP + (size_t)row * 2;
        wts[0] = v0; wts[1] = v1;
        int* idxs = (int*)out + (size_t)ROWS * NEXP + (size_t)ROWS * 2 + (size_t)row * 2;
        idxs[0] = i0; idxs[1] = i1;
    }
}

// ---------------------------------------------------------------------------
extern "C" void kernel_launch(void* const* d_in, const int* in_sizes, int n_in,
                              void* d_out, int out_size, void* d_ws, size_t ws_size,
                              hipStream_t stream)
{
    const float* X  = (const float*)d_in[0];   // [4,4096,2048]
    const float* W1 = (const float*)d_in[1];   // [2048,1024]
    const float* b1 = (const float*)d_in[2];   // [1024]
    const float* W2 = (const float*)d_in[3];   // [1024,8]
    const float* b2 = (const float*)d_in[4];   // [8]
    float* out = (float*)d_out;
    float* Hws = (float*)d_ws;                 // needs ROWS*NDIM*4 = 64 MiB scratch

    dim3 g1(NDIM / BN, ROWS / BM);             // (8, 128)
    router_gemm1<<<g1, 256, 0, stream>>>(X, W1, b1, Hws);
    router_stage2<<<ROWS / 8, 256, 0, stream>>>(Hws, W2, b2, out);
}
